// SocialLSTM_89335319757105
// MI455X (gfx1250) — matrix operand order repaired and checked by reading, not compiled, not run on hardware
//
#include <hip/hip_runtime.h>

// ---------------------------------------------------------------------------
// Social-LSTM fused persistent kernel for MI455X (gfx1250, wave32, WMMA).
//
// Grid: 32 blocks (one per batch). Block: 1024 threads = 32 waves.
// LDS (dynamic, 309 KB of the 320 KB/WGP budget):
//   sHT   : h_t transposed bf16 [256][136 pad]    68 KB (recurrent, persistent)
//   sHs   : social pool for g [128][256] bf16     64 KB (aliased f32 out1/out2)
//   sX    : LSTM input [r|e] [128][576] bf16     144 KB (aliased f32 h2 later)
//   sXY   : positions [128][2] f32                 1 KB
//   sCell : grid-cell id per (i,j) pair u16       32 KB (masks are a partition)
// Cell state c lives in VGPRs per wave for all 64 steps. The gate GEMM runs
// as TWO passes of 8 accumulator tiles (i,g then f,o) to keep peak VGPR
// pressure ~160 and avoid scratch spills at 32 waves/WGP.
// Weights pre-converted to bf16 in d_ws (f32 fallback if ws too small).
// ---------------------------------------------------------------------------

#define HS   256
#define ES   32
#define RS   64
#define GDIR 16
#define NA   128
#define NB   32
#define NT   64
#define KX   576
#define LDT  136   // padded transposed-h stride: 68 dwords -> conflict-free

#define WIH_N (1024 * 576)
#define WHH_N (1024 * 256)
#define WE_N  (32 * 256)

#define OFF_HT   0
#define OFF_HS   69632
#define OFF_X    135168
#define OFF_XY   282624
#define OFF_CELL 283648
#define SMEM_SZ  316416

typedef __bf16 bf16;
typedef __attribute__((ext_vector_type(16))) __bf16 v16bf;
typedef __attribute__((ext_vector_type(8)))  __bf16 v8bf;
typedef __attribute__((ext_vector_type(8)))  float  v8f;
typedef __attribute__((ext_vector_type(4)))  float  v4f;
typedef __attribute__((ext_vector_type(8)))  unsigned short v8u16;

static __device__ __forceinline__ v8f wmma_bf16(v16bf a, v16bf b, v8f c) {
  return __builtin_amdgcn_wmma_f32_16x16x32_bf16(false, a, false, b,
                                                 (short)0, c, false, false);
}

static __device__ __forceinline__ v16bf cat8(v8bf lo, v8bf hi) {
  return __builtin_shufflevector(lo, hi, 0, 1, 2, 3, 4, 5, 6, 7,
                                 8, 9, 10, 11, 12, 13, 14, 15);
}

// A fragment (16x32 bf16) from row-major bf16 LDS tile (base = &M[row0][k0]).
static __device__ __forceinline__ v16bf load_a(const bf16* base, int ld, int lane) {
  const int l = lane & 15, hi = lane >> 4;
  const bf16* p = base + l * ld + hi * 8;
  return cat8(*(const v8bf*)p, *(const v8bf*)(p + 16));
}

// A fragment of h from the TRANSPOSED store (strided; 16 uses per wave-step).
static __device__ __forceinline__ v16bf load_a_hT(const bf16* sHT, int i0, int kh,
                                                  int lane) {
  const int l = lane & 15, hi = lane >> 4;
  const bf16* p = sHT + (kh + hi * 8) * LDT + i0 + l;
  v16bf a;
#pragma unroll
  for (int e = 0; e < 8; ++e) { a[e] = p[e * LDT]; a[8 + e] = p[(16 + e) * LDT]; }
  return a;
}

// B fragment (32x16) of h from the transposed store: 2x ds_load_b128 per lane.
static __device__ __forceinline__ v16bf load_b_hT(const bf16* sHT, int ncol0,
                                                  int kb, int lane) {
  const int n = lane & 15, hi = lane >> 4;
  const bf16* p = sHT + (ncol0 + n) * LDT + kb + hi * 8;
  return cat8(*(const v8bf*)p, *(const v8bf*)(p + 16));
}

// B fragment from pre-converted bf16 weights: 2x global b128, no cvt.
static __device__ __forceinline__ v16bf load_b_wbf(const bf16* __restrict__ W,
                                                   int ldw, int n0, int kb, int lane) {
  const int n = lane & 15, hi = lane >> 4;
  const bf16* p = W + (size_t)(n0 + n) * ldw + kb + hi * 8;
  __builtin_prefetch(p + 32, 0, 0);
  return cat8(*(const v8bf*)p, *(const v8bf*)(p + 16));
}

// f32-weight fallback (vectorized v4f + cvt).
static __device__ __forceinline__ v16bf load_b_wf32(const float* __restrict__ W,
                                                    int ldw, int n0, int kb, int lane) {
  const int n = lane & 15, hi = lane >> 4;
  const float* p = W + (size_t)(n0 + n) * ldw + kb + hi * 8;
  __builtin_prefetch(p + 32, 0, 0);
  const v4f* q = (const v4f*)p;
  v16bf b;
#pragma unroll
  for (int c = 0; c < 4; ++c) {
    v4f f = q[(c & 1) + (c >> 1) * 4];
#pragma unroll
    for (int e = 0; e < 4; ++e) b[c * 4 + e] = (bf16)f[e];
  }
  return b;
}

// Social mask A fragment from u16 cell ids: 2x ds_load_b128, then per element
// just v_cmp_eq_u16 + v_cndmask_b16 (no byte extraction).
static __device__ __forceinline__ v16bf mask_a(const unsigned short* __restrict__ sCell,
                                               int g, int i0, int kb, int lane) {
  const int l = lane & 15, hi = lane >> 4;
  const unsigned short* p = sCell + (i0 + l) * NA + kb + hi * 8;
  const v8u16 c0 = *(const v8u16*)p;
  const v8u16 c1 = *(const v8u16*)(p + 16);
  const unsigned short gs = (unsigned short)g;
  const bf16 one = (bf16)1.0f, zero = (bf16)0.0f;
  v16bf a;
#pragma unroll
  for (int e = 0; e < 8; ++e) {
    a[e]     = (c0[e] == gs) ? one : zero;
    a[8 + e] = (c1[e] == gs) ? one : zero;
  }
  return a;
}

static __device__ __forceinline__ void store_d_bf16(bf16* base, int ld, int lane,
                                                    const v8f& d) {
  bf16* p = base + ((lane >> 4) * 8) * ld + (lane & 15);
#pragma unroll
  for (int r = 0; r < 8; ++r) p[r * ld] = (bf16)d[r];
}

static __device__ __forceinline__ float sigmoidf_(float x) {
  return 1.0f / (1.0f + __expf(-x));
}

// One gate half-pass: 8 output tiles (gates giA, giB x 4 col sub-tiles),
// K = [X (576) | h (256)], weight B double-buffered.
template <bool BFW>
static __device__ __forceinline__ void gate_pass(
    v8f acc[8], const bf16* sX, const bf16* sHT,
    const bf16* __restrict__ bW_ih, const bf16* __restrict__ bW_hh,
    const float* __restrict__ W_ih, const float* __restrict__ W_hh,
    int g_i0, int g_cq, int giA, int giB, int lane) {
  const int nA = giA * 256 + g_cq * 64, nB = giB * 256 + g_cq * 64;
#pragma unroll 1
  for (int ks = 0; ks < 18; ++ks) {            // X part, K = 0..575
    const int kb = ks * 32;
    v16bf aa = load_a(sX + g_i0 * KX + kb, KX, lane);
    v16bf bc = BFW ? load_b_wbf(bW_ih, KX, nA, kb, lane)
                   : load_b_wf32(W_ih, KX, nA, kb, lane);
#pragma unroll
    for (int u = 0; u < 8; ++u) {
      v16bf bn;
      if (u < 7) {
        const int u1 = u + 1;
        const int n0 = (u1 < 4 ? nA : nB) + (u1 & 3) * 16;
        bn = BFW ? load_b_wbf(bW_ih, KX, n0, kb, lane)
                 : load_b_wf32(W_ih, KX, n0, kb, lane);
      }
      acc[u] = wmma_bf16(aa, bc, acc[u]);
      bc = bn;
    }
  }
#pragma unroll 1
  for (int ks = 0; ks < 8; ++ks) {             // h part, K = 576..831
    const int kb = ks * 32;
    v16bf aa = load_a_hT(sHT, g_i0, kb, lane);
    v16bf bc = BFW ? load_b_wbf(bW_hh, HS, nA, kb, lane)
                   : load_b_wf32(W_hh, HS, nA, kb, lane);
#pragma unroll
    for (int u = 0; u < 8; ++u) {
      v16bf bn;
      if (u < 7) {
        const int u1 = u + 1;
        const int n0 = (u1 < 4 ? nA : nB) + (u1 & 3) * 16;
        bn = BFW ? load_b_wbf(bW_hh, HS, n0, kb, lane)
                 : load_b_wf32(W_hh, HS, n0, kb, lane);
      }
      acc[u] = wmma_bf16(aa, bc, acc[u]);
      bc = bn;
    }
  }
}

__global__ void convert_weights_kernel(const float* __restrict__ W_ih,
                                       const float* __restrict__ W_hh,
                                       const float* __restrict__ W_e,
                                       bf16* __restrict__ ws) {
  const int total = WIH_N + WHH_N + WE_N;
  for (int idx = blockIdx.x * blockDim.x + threadIdx.x; idx < total;
       idx += gridDim.x * blockDim.x) {
    float v;
    if (idx < WIH_N)              v = W_ih[idx];
    else if (idx < WIH_N + WHH_N) v = W_hh[idx - WIH_N];
    else                          v = W_e[idx - WIH_N - WHH_N];
    ws[idx] = (bf16)v;
  }
}

template <bool BFW>
__global__ __launch_bounds__(1024, 1)
void social_lstm_kernel(const float* __restrict__ data,
                        const float* __restrict__ W_r,  const float* __restrict__ b_r,
                        const float* __restrict__ W_e,  const float* __restrict__ b_e,
                        const float* __restrict__ W_ih, const float* __restrict__ W_hh,
                        const float* __restrict__ b_ih, const float* __restrict__ b_hh,
                        const float* __restrict__ W_p,  const float* __restrict__ b_p,
                        const float* __restrict__ W_p2, const float* __restrict__ b_p2,
                        const float* __restrict__ W_mu, const float* __restrict__ b_mu,
                        const float* __restrict__ W_sd, const float* __restrict__ b_sd,
                        const float* __restrict__ W_cr, const float* __restrict__ b_cr,
                        const bf16* __restrict__ bW_ih, const bf16* __restrict__ bW_hh,
                        const bf16* __restrict__ bW_e,
                        float* __restrict__ out) {
  extern __shared__ char smem[];
  bf16*  sHT   = (bf16*)(smem + OFF_HT);
  bf16*  sHs   = (bf16*)(smem + OFF_HS);
  bf16*  sX    = (bf16*)(smem + OFF_X);
  float* sXY   = (float*)(smem + OFF_XY);
  unsigned short* sCell = (unsigned short*)(smem + OFF_CELL);
  float* sXf   = (float*)(smem + OFF_X);    // alias sX: h2 f32 after gates
  float* sO1   = (float*)(smem + OFF_HS);   // alias sHs: out1 [128][64]
  float* sO2   = sO1 + NA * 64;             //            out2 [128][32]

  const int tid  = threadIdx.x;
  const int lane = tid & 31;
  const int wave = tid >> 5;
  const int b    = blockIdx.x;

  const int g_i0 = (wave >> 2) * 16;  // fixed wave ownership -> c stays in VGPRs
  const int g_cq = wave & 3;
  const int lnN  = lane & 15;
  const int lnHR = (lane >> 4) * 8;

  float creg[4][8];
#pragma unroll
  for (int s = 0; s < 4; ++s)
#pragma unroll
    for (int r = 0; r < 8; ++r) creg[s][r] = 0.0f;

  for (int idx = tid; idx < HS * LDT; idx += 1024) sHT[idx] = (bf16)0.0f;
  __syncthreads();

  const int sdOff  = NB * NA * NT * 2;
  const int crBase = sdOff * 2;

#pragma unroll 1
  for (int t = 0; t < NT; ++t) {
    // ---- Phase 1: positions, r = relu(xt@W_r^T+b_r), grid-cell ids --------
#pragma unroll 1
    for (int idx = tid; idx < NA * 2; idx += 1024) {
      const int a = idx >> 1, k = idx & 1;
      sXY[idx] = data[(((size_t)b * NA + a) * NT + t) * 2 + k];
    }
    __syncthreads();
#pragma unroll 1
    for (int idx = tid; idx < NA * RS; idx += 1024) {
      const int a = idx >> 6, j = idx & 63;
      const float v = W_r[2 * j] * sXY[2 * a] + W_r[2 * j + 1] * sXY[2 * a + 1] + b_r[j];
      sX[a * KX + j] = (bf16)fmaxf(v, 0.0f);
    }
    // The 16 direction masks partition the 32x32 box: cell = 4*ybin + xbin.
#pragma unroll 1
    for (int idx = tid; idx < NA * NA; idx += 1024) {
      const int i = idx >> 7, j = idx & 127;
      const float xi = sXY[2 * i], yi = sXY[2 * i + 1];
      const float xj = sXY[2 * j], yj = sXY[2 * j + 1];
      const float dx = xi - xj, dy = yi - yj;
      unsigned short c = 0xFFFF;
      if (xi >= 0.0f && xj >= 0.0f && i != j &&
          dx >= -16.0f && dx <= 16.0f && dy >= -16.0f && dy <= 16.0f) {
        int q = (int)floorf((16.0f - dx) * 0.125f); q = q > 3 ? 3 : q;
        int p = (int)floorf((dy + 16.0f) * 0.125f); p = p > 3 ? 3 : p;
        c = (unsigned short)(p * 4 + q);
      }
      sCell[idx] = c;
    }
    __syncthreads();

    // ---- Phase 2: social pooling + e projection per grid direction --------
#pragma unroll 1
    for (int g = 0; g < GDIR; ++g) {
      // 2a. Hs_g = mask_g @ h; one h-col tile per pass, B frags preloaded.
      {
        v16bf am[4];
#pragma unroll
        for (int kv = 0; kv < 4; ++kv) am[kv] = mask_a(sCell, g, g_i0, kv * 32, lane);
#pragma unroll
        for (int q = 0; q < 4; ++q) {
          const int hcol = (wave & 3) * 4 + q;
          v16bf bb[4];
#pragma unroll
          for (int kv = 0; kv < 4; ++kv)
            bb[kv] = load_b_hT(sHT, hcol * 16, kv * 32, lane);
          v8f acc{};
#pragma unroll
          for (int kv = 0; kv < 4; ++kv) acc = wmma_bf16(am[kv], bb[kv], acc);
          store_d_bf16(sHs + g_i0 * HS + hcol * 16, HS, lane, acc);
        }
      }
      __syncthreads();

      // 2b. e_g = relu(Hs_g @ W_e^T + b_e), exact torch reshape scatter.
      if (wave < 16) {
        const int eit = wave >> 1, ct = wave & 1;
        v8f e0{}, e1{};
#pragma unroll 1
        for (int ks = 0; ks < 8; ks += 2) {
          v16bf a0 = load_a(sHs + (eit * 16) * HS + ks * 32, HS, lane);
          v16bf a1 = load_a(sHs + (eit * 16) * HS + (ks + 1) * 32, HS, lane);
          v16bf q0 = BFW ? load_b_wbf(bW_e, HS, ct * 16, ks * 32, lane)
                         : load_b_wf32(W_e, HS, ct * 16, ks * 32, lane);
          v16bf q1 = BFW ? load_b_wbf(bW_e, HS, ct * 16, (ks + 1) * 32, lane)
                         : load_b_wf32(W_e, HS, ct * 16, (ks + 1) * 32, lane);
          e0 = wmma_bf16(a0, q0, e0);
          e1 = wmma_bf16(a1, q1, e1);
        }
        const v8f acc = e0 + e1;
        const int ecol = ct * 16 + lnN;
        const float be = b_e[ecol];
#pragma unroll
        for (int r = 0; r < 8; ++r) {
          const int i = eit * 16 + lnHR + r;
          const float v = fmaxf(acc[r] + be, 0.0f);
          sX[(g * 8 + (i >> 4)) * KX + RS + (i & 15) * 32 + ecol] = (bf16)v;
        }
      }
      __syncthreads();
    }

    // ---- Phase 3a: gates i,g -> p = sigmoid(i)*tanh(g) --------------------
    float preg[4][8];
    {
      v8f acc[8];
      { const v8f z{};
#pragma unroll
        for (int u = 0; u < 8; ++u) acc[u] = z; }
      gate_pass<BFW>(acc, sX, sHT, bW_ih, bW_hh, W_ih, W_hh,
                     g_i0, g_cq, /*giA=*/0, /*giB=*/2, lane);
#pragma unroll
      for (int s = 0; s < 4; ++s) {
        const int colh = g_cq * 64 + s * 16 + lnN;
        const float bi = b_ih[colh]       + b_hh[colh];
        const float bg = b_ih[512 + colh] + b_hh[512 + colh];
#pragma unroll
        for (int r = 0; r < 8; ++r)
          preg[s][r] = sigmoidf_(acc[s][r] + bi) * tanhf(acc[4 + s][r] + bg);
      }
    }

    // ---- Phase 3b: gates f,o -> c2, h2 ------------------------------------
    {
      v8f acc[8];
      { const v8f z{};
#pragma unroll
        for (int u = 0; u < 8; ++u) acc[u] = z; }
      gate_pass<BFW>(acc, sX, sHT, bW_ih, bW_hh, W_ih, W_hh,
                     g_i0, g_cq, /*giA=*/1, /*giB=*/3, lane);
      __syncthreads();   // all waves done reading sX / sHT before rewrite
#pragma unroll
      for (int s = 0; s < 4; ++s) {
        const int colh = g_cq * 64 + s * 16 + lnN;
        const float bf = b_ih[256 + colh] + b_hh[256 + colh];
        const float bo = b_ih[768 + colh] + b_hh[768 + colh];
#pragma unroll
        for (int r = 0; r < 8; ++r) {
          const int row = g_i0 + lnHR + r;
          const float fg = sigmoidf_(acc[s][r] + bf);
          const float og = sigmoidf_(acc[4 + s][r] + bo);
          const float c2 = fg * creg[s][r] + preg[s][r];
          const float h2 = og * tanhf(c2);
          creg[s][r] = c2;
          sHT[colh * LDT + row] = (bf16)h2;
          sXf[row * HS + colh]  = h2;
        }
      }
    }
    __syncthreads();

    // ---- Phase 5: output head ---------------------------------------------
#pragma unroll 1
    for (int idx = tid; idx < NA * 64; idx += 1024) {
      const int row = idx >> 6, o = idx & 63;
      const float* wp = W_p + o * 320;
      const float x = sXY[2 * row], y = sXY[2 * row + 1];
      float sum = b_p[o];
#pragma unroll 4
      for (int k = 0; k < RS; ++k) {
        const float rk = fmaxf(W_r[2 * k] * x + W_r[2 * k + 1] * y + b_r[k], 0.0f);
        sum += wp[k] * rk;
      }
      const float* h2p = sXf + row * HS;
#pragma unroll 4
      for (int k = 0; k < HS; ++k) sum += wp[RS + k] * h2p[k];
      sO1[idx] = sum;
    }
    __syncthreads();
#pragma unroll 1
    for (int idx = tid; idx < NA * 32; idx += 1024) {
      const int row = idx >> 5, o = idx & 31;
      const float* w2 = W_p2 + o * 64;
      const float* o1 = sO1 + row * 64;
      float sum = b_p2[o];
#pragma unroll 4
      for (int k = 0; k < 64; ++k) sum += w2[k] * o1[k];
      sO2[idx] = sum;
    }
    __syncthreads();
    if (tid < NA * 5) {
      const int row = tid / 5, q = tid % 5;
      const float* o2 = sO2 + row * 32;
      const int base2 = ((b * NA + row) * NT + t);
      if (q < 2) {
        float sum = b_mu[q];
        for (int k = 0; k < 32; ++k) sum += W_mu[q * 32 + k] * o2[k];
        out[base2 * 2 + q] = sum;
      } else if (q < 4) {
        const int qq = q - 2;
        float sum = b_sd[qq];
        for (int k = 0; k < 32; ++k) sum += W_sd[qq * 32 + k] * o2[k];
        out[sdOff + base2 * 2 + qq] = __expf(sum);
      } else {
        float sum = b_cr[0];
        for (int k = 0; k < 32; ++k) sum += W_cr[k] * o2[k];
        out[crBase + base2] = tanhf(sum);
      }
    }
    __syncthreads();
  }
}

extern "C" void kernel_launch(void* const* d_in, const int* in_sizes, int n_in,
                              void* d_out, int out_size, void* d_ws, size_t ws_size,
                              hipStream_t stream) {
  (void)in_sizes; (void)n_in; (void)out_size;
  const float* data = (const float*)d_in[0];
  const float* W_r  = (const float*)d_in[2];
  const float* b_r  = (const float*)d_in[3];
  const float* W_e  = (const float*)d_in[4];
  const float* b_e  = (const float*)d_in[5];
  const float* W_ih = (const float*)d_in[6];
  const float* W_hh = (const float*)d_in[7];
  const float* b_ih = (const float*)d_in[8];
  const float* b_hh = (const float*)d_in[9];
  const float* W_p  = (const float*)d_in[10];
  const float* b_p  = (const float*)d_in[11];
  const float* W_p2 = (const float*)d_in[12];
  const float* b_p2 = (const float*)d_in[13];
  const float* W_mu = (const float*)d_in[14];
  const float* b_mu = (const float*)d_in[15];
  const float* W_sd = (const float*)d_in[16];
  const float* b_sd = (const float*)d_in[17];
  const float* W_cr = (const float*)d_in[18];
  const float* b_cr = (const float*)d_in[19];
  float* out = (float*)d_out;

  const size_t needWs = (size_t)(WIH_N + WHH_N + WE_N) * sizeof(bf16);
  const bool useBf = (d_ws != nullptr) && (ws_size >= needWs);

  static bool attrSet = false;
  if (!attrSet) {
    hipFuncSetAttribute((const void*)social_lstm_kernel<true>,
                        hipFuncAttributeMaxDynamicSharedMemorySize, SMEM_SZ);
    hipFuncSetAttribute((const void*)social_lstm_kernel<false>,
                        hipFuncAttributeMaxDynamicSharedMemorySize, SMEM_SZ);
    attrSet = true;
  }

  if (useBf) {
    bf16* ws = (bf16*)d_ws;
    convert_weights_kernel<<<dim3(512), dim3(256), 0, stream>>>(W_ih, W_hh, W_e, ws);
    social_lstm_kernel<true><<<dim3(NB), dim3(1024), SMEM_SZ, stream>>>(
        data, W_r, b_r, W_e, b_e, W_ih, W_hh, b_ih, b_hh,
        W_p, b_p, W_p2, b_p2, W_mu, b_mu, W_sd, b_sd, W_cr, b_cr,
        ws, ws + WIH_N, ws + WIH_N + WHH_N, out);
  } else {
    social_lstm_kernel<false><<<dim3(NB), dim3(1024), SMEM_SZ, stream>>>(
        data, W_r, b_r, W_e, b_e, W_ih, W_hh, b_ih, b_hh,
        W_p, b_p, W_p2, b_p2, W_mu, b_mu, W_sd, b_sd, W_cr, b_cr,
        nullptr, nullptr, nullptr, out);
  }
}
